// VectorQuantizer_49134425866715
// MI455X (gfx1250) — matrix operand order repaired and checked by reading, not compiled, not run on hardware
//
#include <hip/hip_runtime.h>

// ---------------------------------------------------------------------------
// VectorQuantizer for MI455X (gfx1250).
//   d' = ||e||^2 - 2 z.e^T  (||z||^2 dropped: constant per row for argmin)
//   GEMM (32768x1024x64, 4.3 GFLOP) via v_wmma_f32_16x16x32_bf16 with the
//   bf16 hi/lo 3-term split (hi*hi + hi*lo + lo*hi) ~ fp32 precision.
//   Two independent accumulators (one per K-step) shorten the WMMA RAW chain.
//   8 waves/WG so the 256 KB split codebook is swept by fewer workgroups
//   (halves L2->WGP traffic vs 4 waves/WG).
//   loss = 1.25 * mean((z_q - z)^2), z_q_1 == z_q numerically.
// ---------------------------------------------------------------------------

typedef __attribute__((ext_vector_type(16))) __bf16 v16bf;
typedef __attribute__((ext_vector_type(8)))  float  v8f;
typedef __attribute__((ext_vector_type(4)))  unsigned int u32x4;

#define N_E        1024
#define E_DIM      64
#define HWSZ       1024                 // 32*32
#define N_ROWS     32768                // 32*32*32
#define N_TILES    (N_E / 16)           // 64 column tiles of 16 codes
#define WAVES_WG   8
#define WG_SIZE    (WAVES_WG * 32)      // 256
#define N_WG       (N_ROWS / (WAVES_WG * 16))   // 256
#define N_PART     (N_WG * WAVES_WG)            // 2048
#define INV_COUNT  (1.0f / 2097152.0f)          // 1 / (32*64*32*32)

union AFrag { v16bf v; unsigned short s[16]; };
union BFrag { v16bf v; u32x4 q[2]; };

__device__ __forceinline__ unsigned short f32_to_bf16_rne(float x) {
    unsigned u = __float_as_uint(x);
    unsigned r = u + 0x7FFFu + ((u >> 16) & 1u);
    return (unsigned short)(r >> 16);
}

__device__ __forceinline__ void bf16_split(float x, unsigned short& hi,
                                           unsigned short& lo) {
    unsigned short h = f32_to_bf16_rne(x);
    float hf = __uint_as_float(((unsigned)h) << 16);
    hi = h;
    lo = f32_to_bf16_rne(x - hf);
}

// --- kernel 1: codebook norms + bf16 hi/lo split ---------------------------
__global__ void vq_prep_kernel(const float* __restrict__ cb,
                               unsigned short* __restrict__ cb_hi,
                               unsigned short* __restrict__ cb_lo,
                               float* __restrict__ enorm) {
    int r = blockIdx.x * blockDim.x + threadIdx.x;
    if (r >= N_E) return;
    const float* row = cb + (size_t)r * E_DIM;
    float nrm = 0.0f;
    #pragma unroll 8
    for (int k = 0; k < E_DIM; ++k) {
        float x = row[k];
        nrm += x * x;
        unsigned short h, l;
        bf16_split(x, h, l);
        cb_hi[(size_t)r * E_DIM + k] = h;
        cb_lo[(size_t)r * E_DIM + k] = l;
    }
    enorm[r] = nrm;
}

// --- kernel 2: WMMA distance + argmin + gather + SSE partials --------------
__global__ void __launch_bounds__(WG_SIZE)
vq_main_kernel(const float* __restrict__ z,
               const float* __restrict__ cb,
               const unsigned short* __restrict__ cb_hi,
               const unsigned short* __restrict__ cb_lo,
               const float* __restrict__ enorm,
               float* __restrict__ outq,        // d_out + 1, [B,C,H,W] flat
               float* __restrict__ partials) {
    __shared__ int s_idx[WAVES_WG * 16];

    const int lane = threadIdx.x & 31;
    const int wave = threadIdx.x >> 5;
    const int m    = lane & 15;          // row within 16-row tile / col within N-tile
    const int h    = lane >> 4;          // lane half

    const int row_base = blockIdx.x * (WAVES_WG * 16) + wave * 16;
    const int b        = row_base >> 10;           // batch image (128 | 1024)
    const int hw       = (row_base & 1023) + m;    // position in HxW plane
    const float* zb    = z + (size_t)b * E_DIM * HWSZ;

    // ---- Build A fragments (this lane = row m, lane-half h) ----
    // A k order chosen here (matches B): elem j -> K = kstep*32 + 16h + j
    AFrag a_hi[2], a_lo[2];
    #pragma unroll
    for (int s = 0; s < 2; ++s) {
        #pragma unroll
        for (int j = 0; j < 16; ++j) {
            int K = s * 32 + 16 * h + j;
            float x = zb[(size_t)K * HWSZ + hw];
            bf16_split(x, a_hi[s].s[j], a_lo[s].s[j]);
        }
    }

    // ---- Sweep 64 column tiles of the codebook ----
    float minv[8];
    int   mini[8];
    #pragma unroll
    for (int r = 0; r < 8; ++r) { minv[r] = 3.4e38f; mini[r] = 0; }

    for (int t = 0; t < N_TILES; ++t) {
        const int n = t * 16 + m;        // this lane's codebook column
        // B k order chosen here (matches A): elem j -> K = kstep*32 + 16h + j ; 32B-aligned rows.
        BFrag bh0, bl0, bh1, bl1;
        {
            const size_t base = (size_t)n * E_DIM + 16 * h;
            const u32x4* ph0 = (const u32x4*)(cb_hi + base);
            const u32x4* pl0 = (const u32x4*)(cb_lo + base);
            const u32x4* ph1 = (const u32x4*)(cb_hi + base + 32);
            const u32x4* pl1 = (const u32x4*)(cb_lo + base + 32);
            bh0.q[0] = ph0[0]; bh0.q[1] = ph0[1];
            bl0.q[0] = pl0[0]; bl0.q[1] = pl0[1];
            bh1.q[0] = ph1[0]; bh1.q[1] = ph1[1];
            bl1.q[0] = pl1[0]; bl1.q[1] = pl1[1];
        }
        const float en = enorm[n];

        // Two independent accumulator chains (K-step 0 / K-step 1).
        v8f c0 = {};
        v8f c1 = {};
        c0 = __builtin_amdgcn_wmma_f32_16x16x32_bf16(false, a_lo[0].v, false, bh0.v, (short)0, c0, false, false);
        c1 = __builtin_amdgcn_wmma_f32_16x16x32_bf16(false, a_lo[1].v, false, bh1.v, (short)0, c1, false, false);
        c0 = __builtin_amdgcn_wmma_f32_16x16x32_bf16(false, a_hi[0].v, false, bl0.v, (short)0, c0, false, false);
        c1 = __builtin_amdgcn_wmma_f32_16x16x32_bf16(false, a_hi[1].v, false, bl1.v, (short)0, c1, false, false);
        c0 = __builtin_amdgcn_wmma_f32_16x16x32_bf16(false, a_hi[0].v, false, bh0.v, (short)0, c0, false, false);
        c1 = __builtin_amdgcn_wmma_f32_16x16x32_bf16(false, a_hi[1].v, false, bh1.v, (short)0, c1, false, false);

        #pragma unroll
        for (int r = 0; r < 8; ++r) {
            float d = en - 2.0f * (c0[r] + c1[r]);   // C VGPR r = row r + 8h
            if (d < minv[r]) { minv[r] = d; mini[r] = n; }  // earliest-n tiebreak
        }
    }

    // ---- Cross-lane argmin within each 16-lane half (first-occurrence) ----
    #pragma unroll
    for (int sh = 0; sh < 4; ++sh) {
        const int mask = 1 << sh;
        #pragma unroll
        for (int r = 0; r < 8; ++r) {
            float ov = __shfl_xor(minv[r], mask, 16);
            int   oi = __shfl_xor(mini[r], mask, 16);
            if (ov < minv[r] || (ov == minv[r] && oi < mini[r])) {
                minv[r] = ov; mini[r] = oi;
            }
        }
    }
    // lanes 0..15 now hold rows 0..7 (r=0..7); lanes 16..31 hold rows 8..15.
    if ((lane & 15) == 0) {
        #pragma unroll
        for (int r = 0; r < 8; ++r) s_idx[wave * 16 + h * 8 + r] = mini[r];
    }
    __syncthreads();

    // ---- Gather winning codes -> z_q, accumulate SSE ----
    const int myidx = s_idx[wave * 16 + m];
    const float* crow = cb + (size_t)myidx * E_DIM;
    float* outb = outq + (size_t)b * E_DIM * HWSZ;
    float acc = 0.0f;
    #pragma unroll 8
    for (int k0 = 0; k0 < 32; ++k0) {
        const int k = 2 * k0 + h;                 // half-wave -> 64B contiguous
        const float cv = crow[k];
        const size_t off = (size_t)k * HWSZ + hw;
        const float zv = zb[off];
        outb[off] = cv;
        const float d = cv - zv;
        acc += d * d;
    }
    // wave reduction (fixed order within wave -> deterministic partial)
    #pragma unroll
    for (int mask = 16; mask >= 1; mask >>= 1) acc += __shfl_xor(acc, mask, 32);
    if (lane == 0) partials[blockIdx.x * WAVES_WG + wave] = acc;
}

// --- kernel 3: deterministic fixed-order reduction -> loss -----------------
__global__ void vq_reduce_kernel(const float* __restrict__ partials,
                                 float* __restrict__ out) {
    __shared__ float sm[256];
    const int t = threadIdx.x;
    float s = 0.0f;
    for (int i = t; i < N_PART; i += 256) s += partials[i];
    sm[t] = s;
    __syncthreads();
    for (int w = 128; w > 0; w >>= 1) {
        if (t < w) sm[t] += sm[t + w];
        __syncthreads();
    }
    if (t == 0) out[0] = 1.25f * sm[0] * INV_COUNT;  // (1 + BETA) * mean
}

extern "C" void kernel_launch(void* const* d_in, const int* in_sizes, int n_in,
                              void* d_out, int out_size, void* d_ws, size_t ws_size,
                              hipStream_t stream) {
    (void)in_sizes; (void)n_in; (void)out_size; (void)ws_size;
    const float* z  = (const float*)d_in[0];   // [32,64,32,32] fp32
    const float* cb = (const float*)d_in[1];   // [1024,64] fp32
    float* out = (float*)d_out;                // [0]=loss, [1..]=z_q flat

    char* ws = (char*)d_ws;
    unsigned short* cb_hi = (unsigned short*)(ws);             // 128 KB
    unsigned short* cb_lo = (unsigned short*)(ws + 131072);    // 128 KB
    float* enorm    = (float*)(ws + 262144);                   //   4 KB
    float* partials = (float*)(ws + 266240);                   //   8 KB

    vq_prep_kernel<<<(N_E + 255) / 256, 256, 0, stream>>>(cb, cb_hi, cb_lo, enorm);
    vq_main_kernel<<<N_WG, WG_SIZE, 0, stream>>>(z, cb, cb_hi, cb_lo, enorm,
                                                 out + 1, partials);
    vq_reduce_kernel<<<1, 256, 0, stream>>>(partials, out);
}
